// MultiheadAttention_79053168050795
// MI455X (gfx1250) — matrix operand order repaired and checked by reading, not compiled, not run on hardware
//
#include <hip/hip_runtime.h>

typedef __bf16 bf16_t;
typedef __attribute__((ext_vector_type(16))) __bf16 v16bf;
typedef __attribute__((ext_vector_type(8)))  __bf16 v8bf;
typedef __attribute__((ext_vector_type(8)))  float  v8f;

#define NB    2
#define SEQ   2048
#define DM    1024
#define NH    16
#define DEP   64
#define MTOT  (NB * SEQ)

union V16 { v16bf v; v8bf h[2]; };

static __device__ __forceinline__ v8f wmma_bf16(v16bf a, v16bf b, v8f c) {
  return __builtin_amdgcn_wmma_f32_16x16x32_bf16(false, a, false, b, (short)0, c,
                                                 false, false);
}

// Async 16-byte global -> LDS copy (per active lane), tracked by ASYNCcnt.
static __device__ __forceinline__ void async_copy_b128(const void* gsrc, void* lds_dst) {
  const unsigned loff = (unsigned)(size_t)lds_dst;  // low 32 bits = LDS offset
  asm volatile("global_load_async_to_lds_b128 %0, %1, off"
               :: "v"(loff), "v"(gsrc)
               : "memory");
}
static __device__ __forceinline__ void wait_async0() {
  asm volatile("s_wait_asynccnt 0x0" ::: "memory");
}

// ---------------------------------------------------------------------------
// GEMM: C[M,N] = A(fp32)[M,K] @ W(fp32)[K,N], out bf16 head-split [B,H,S,64].
// Block 256 thr = 8 waves; tile 128(M) x 128(N); wave = 32x64 (8 WMMA/K-step).
// ---------------------------------------------------------------------------
__global__ __launch_bounds__(256) void gemm_qkv(const float* __restrict__ A,
                                                const float* __restrict__ W,
                                                bf16_t* __restrict__ out) {
  __shared__ bf16_t As[128][40];   // [m][k], +8 pad
  __shared__ bf16_t Bt[128][40];   // [n][k] transposed, +8 pad

  const int tid = threadIdx.x;
  const int wave = tid >> 5, lane = tid & 31;
  const int lh = lane >> 4, l16 = lane & 15;
  const int mw = wave & 3, nw = wave >> 2;
  const int m0 = blockIdx.x * 128, n0 = blockIdx.y * 128;

  v8f acc[2][4] = {};

  const int ar = tid >> 3, ac = (tid & 7) * 4;     // A: 8 thr/row, float4
  const int br = tid >> 5, bc = (tid & 31) * 4;    // W: 32 thr/row, float4

  for (int k0 = 0; k0 < DM; k0 += 32) {
    __syncthreads();
#pragma unroll
    for (int rr = 0; rr < 128; rr += 32) {
      float4 f = *reinterpret_cast<const float4*>(
          A + (size_t)(m0 + ar + rr) * DM + k0 + ac);
      As[ar + rr][ac + 0] = (bf16_t)f.x;
      As[ar + rr][ac + 1] = (bf16_t)f.y;
      As[ar + rr][ac + 2] = (bf16_t)f.z;
      As[ar + rr][ac + 3] = (bf16_t)f.w;
    }
#pragma unroll
    for (int rr = 0; rr < 32; rr += 8) {
      float4 f = *reinterpret_cast<const float4*>(
          W + (size_t)(k0 + br + rr) * DM + n0 + bc);
      Bt[bc + 0][br + rr] = (bf16_t)f.x;
      Bt[bc + 1][br + rr] = (bf16_t)f.y;
      Bt[bc + 2][br + rr] = (bf16_t)f.z;
      Bt[bc + 3][br + rr] = (bf16_t)f.w;
    }
    if (k0 + 32 < DM) {  // prefetch next K-step lines
      __builtin_prefetch(A + (size_t)(m0 + ar) * DM + k0 + 32 + ac, 0, 3);
      __builtin_prefetch(W + (size_t)(k0 + 32 + br) * DM + n0 + bc, 0, 3);
    }
    __syncthreads();

    V16 af[2];
#pragma unroll
    for (int i = 0; i < 2; ++i) {
      const int row = mw * 32 + i * 16 + l16;
      af[i].h[0] = *reinterpret_cast<const v8bf*>(&As[row][8 * lh]);
      af[i].h[1] = *reinterpret_cast<const v8bf*>(&As[row][16 + 8 * lh]);
    }
#pragma unroll
    for (int g = 0; g < 4; ++g) {
      V16 bfm;
      bfm.h[0] = *reinterpret_cast<const v8bf*>(&Bt[nw * 64 + g * 16 + l16][8 * lh]);
      bfm.h[1] = *reinterpret_cast<const v8bf*>(&Bt[nw * 64 + g * 16 + l16][16 + 8 * lh]);
#pragma unroll
      for (int i = 0; i < 2; ++i)
        acc[i][g] = wmma_bf16(af[i].v, bfm.v, acc[i][g]);
    }
  }

#pragma unroll
  for (int i = 0; i < 2; ++i) {
#pragma unroll
    for (int g = 0; g < 4; ++g) {
      const int n = n0 + nw * 64 + g * 16 + l16;
      const int h = n >> 6, d = n & 63;
#pragma unroll
      for (int r = 0; r < 8; ++r) {
        const int m = m0 + mw * 32 + i * 16 + r + 8 * lh;
        const int b = m >> 11, s = m & (SEQ - 1);
        out[((size_t)(b * NH + h) * SEQ + s) * DEP + d] = (bf16_t)acc[i][g][r];
      }
    }
  }
}

// ---------------------------------------------------------------------------
// GEMM: out(fp32)[M,N] = A(bf16)[M,K] @ W(fp32)[K,N]   (final projection).
// A tile staged via GLOBAL_LOAD_ASYNC_TO_LDS_B128 (no conversion needed).
// ---------------------------------------------------------------------------
__global__ __launch_bounds__(256) void gemm_out(const bf16_t* __restrict__ A,
                                                const float* __restrict__ W,
                                                float* __restrict__ out) {
  __shared__ bf16_t As[128][40];
  __shared__ bf16_t Bt[128][40];

  const int tid = threadIdx.x;
  const int wave = tid >> 5, lane = tid & 31;
  const int lh = lane >> 4, l16 = lane & 15;
  const int mw = wave & 3, nw = wave >> 2;
  const int m0 = blockIdx.x * 128, n0 = blockIdx.y * 128;

  v8f acc[2][4] = {};

  const int ar = tid >> 1, ac = (tid & 1) * 16;    // A: 2 thr/row, 2x16B async
  const int br = tid >> 5, bc = (tid & 31) * 4;    // W: 32 thr/row, float4

  for (int k0 = 0; k0 < DM; k0 += 32) {
    __syncthreads();
    // async stage A tile 128x32 bf16 directly into LDS
    async_copy_b128(A + (size_t)(m0 + ar) * DM + k0 + ac, &As[ar][ac]);
    async_copy_b128(A + (size_t)(m0 + ar) * DM + k0 + ac + 8, &As[ar][ac + 8]);
    // stage W tile 32x128 fp32 -> bf16 transposed (overlaps with async copy)
#pragma unroll
    for (int rr = 0; rr < 32; rr += 8) {
      float4 f = *reinterpret_cast<const float4*>(
          W + (size_t)(k0 + br + rr) * DM + n0 + bc);
      Bt[bc + 0][br + rr] = (bf16_t)f.x;
      Bt[bc + 1][br + rr] = (bf16_t)f.y;
      Bt[bc + 2][br + rr] = (bf16_t)f.z;
      Bt[bc + 3][br + rr] = (bf16_t)f.w;
    }
    if (k0 + 32 < DM) {
      __builtin_prefetch(W + (size_t)(k0 + 32 + br) * DM + n0 + bc, 0, 3);
    }
    wait_async0();
    __syncthreads();

    V16 af[2];
#pragma unroll
    for (int i = 0; i < 2; ++i) {
      const int row = mw * 32 + i * 16 + l16;
      af[i].h[0] = *reinterpret_cast<const v8bf*>(&As[row][8 * lh]);
      af[i].h[1] = *reinterpret_cast<const v8bf*>(&As[row][16 + 8 * lh]);
    }
#pragma unroll
    for (int g = 0; g < 4; ++g) {
      V16 bfm;
      bfm.h[0] = *reinterpret_cast<const v8bf*>(&Bt[nw * 64 + g * 16 + l16][8 * lh]);
      bfm.h[1] = *reinterpret_cast<const v8bf*>(&Bt[nw * 64 + g * 16 + l16][16 + 8 * lh]);
#pragma unroll
      for (int i = 0; i < 2; ++i)
        acc[i][g] = wmma_bf16(af[i].v, bfm.v, acc[i][g]);
    }
  }

#pragma unroll
  for (int i = 0; i < 2; ++i) {
#pragma unroll
    for (int g = 0; g < 4; ++g) {
      const int n = n0 + nw * 64 + g * 16 + l16;
#pragma unroll
      for (int r = 0; r < 8; ++r) {
        const int m = m0 + mw * 32 + i * 16 + r + 8 * lh;
        out[(size_t)m * DM + n] = acc[i][g][r];
      }
    }
  }
}

// ---------------------------------------------------------------------------
// Flash attention. Block = 128 query rows of one (b,h); wave = 16 rows.
// K tile staged via async global->LDS; V transposed through VGPRs.
// ---------------------------------------------------------------------------
__global__ __launch_bounds__(256) void attn(const bf16_t* __restrict__ Qh,
                                            const bf16_t* __restrict__ Kh,
                                            const bf16_t* __restrict__ Vh,
                                            const float* __restrict__ mask,
                                            bf16_t* __restrict__ O) {
  __shared__ bf16_t Kt[32][72];      // [key][depth 64], +8 pad
  __shared__ bf16_t Vt[64][40];      // [depth][key 32], +8 pad
  __shared__ bf16_t Pl[8][16][40];   // per-wave P: [wave][row][key 32], +8 pad

  const int tid = threadIdx.x;
  const int wave = tid >> 5, lane = tid & 31;
  const int lh = lane >> 4, l16 = lane & 15;
  const int q0 = blockIdx.x * 128 + wave * 16;
  const int h = blockIdx.y;
  const int b = blockIdx.z;

  const bf16_t* Qb = Qh + (size_t)(b * NH + h) * SEQ * DEP;
  const bf16_t* Kb = Kh + (size_t)(b * NH + h) * SEQ * DEP;
  const bf16_t* Vb = Vh + (size_t)(b * NH + h) * SEQ * DEP;

  // Q fragments (16 rows x depth 64 -> two 16x32 A fragments), kept in regs
  V16 qf[2];
  {
    const bf16_t* qrow = Qb + (size_t)(q0 + l16) * DEP;
#pragma unroll
    for (int j = 0; j < 2; ++j) {
      qf[j].h[0] = *reinterpret_cast<const v8bf*>(qrow + 32 * j + 8 * lh);
      qf[j].h[1] = *reinterpret_cast<const v8bf*>(qrow + 32 * j + 16 + 8 * lh);
    }
  }

  float mrow[8], lrow[8];
#pragma unroll
  for (int r = 0; r < 8; ++r) { mrow[r] = -1e30f; lrow[r] = 0.0f; }
  v8f acc[4] = {};

  const int cr = tid >> 3, cc = (tid & 7) * 8;  // cooperative tile loaders

  for (int k0 = 0; k0 < SEQ; k0 += 32) {
    __syncthreads();
    // K tile [32 x 64]: async global -> LDS (16B per lane)
    async_copy_b128(Kb + (size_t)(k0 + cr) * DEP + cc, &Kt[cr][cc]);
    // V tile transposed [64 x 32] through VGPRs (overlaps with the async copy)
    {
      v8bf vv = *reinterpret_cast<const v8bf*>(Vb + (size_t)(k0 + cr) * DEP + cc);
#pragma unroll
      for (int i = 0; i < 8; ++i) Vt[cc + i][cr] = vv[i];
    }
    wait_async0();
    __syncthreads();

    // scores: 16 q-rows x 32 keys  (2 key-groups x 2 depth-chunks of WMMA)
    v8f sc[2] = {};
#pragma unroll
    for (int ng = 0; ng < 2; ++ng) {
#pragma unroll
      for (int kc = 0; kc < 2; ++kc) {
        V16 bk;  // B fragment: lane = key col, K = depth 32kc + table order (8*lh, 16+8*lh)
        bk.h[0] = *reinterpret_cast<const v8bf*>(&Kt[ng * 16 + l16][kc * 32 + 8 * lh]);
        bk.h[1] = *reinterpret_cast<const v8bf*>(&Kt[ng * 16 + l16][kc * 32 + 16 + 8 * lh]);
        sc[ng] = wmma_bf16(qf[kc].v, bk.v, sc[ng]);
      }
    }

    // scale + additive mask (mask broadcast over q-rows and heads)
    const float mk0 = -1e9f * mask[(size_t)b * SEQ + k0 + l16];
    const float mk1 = -1e9f * mask[(size_t)b * SEQ + k0 + 16 + l16];
#pragma unroll
    for (int r = 0; r < 8; ++r) {
      sc[0][r] = sc[0][r] * 0.125f + mk0;
      sc[1][r] = sc[1][r] * 0.125f + mk1;
    }

    // online softmax (row = r + 8*lh lives across 16 lanes of this half)
    float alpha[8];
#pragma unroll
    for (int r = 0; r < 8; ++r) {
      float v = fmaxf(sc[0][r], sc[1][r]);
      v = fmaxf(v, __shfl_xor(v, 1, 32));
      v = fmaxf(v, __shfl_xor(v, 2, 32));
      v = fmaxf(v, __shfl_xor(v, 4, 32));
      v = fmaxf(v, __shfl_xor(v, 8, 32));
      const float mnew = fmaxf(mrow[r], v);
      alpha[r] = __expf(mrow[r] - mnew);
      mrow[r] = mnew;
      const float p0 = __expf(sc[0][r] - mnew);
      const float p1 = __expf(sc[1][r] - mnew);
      sc[0][r] = p0;
      sc[1][r] = p1;
      float ps = p0 + p1;
      ps += __shfl_xor(ps, 1, 32);
      ps += __shfl_xor(ps, 2, 32);
      ps += __shfl_xor(ps, 4, 32);
      ps += __shfl_xor(ps, 8, 32);
      lrow[r] = lrow[r] * alpha[r] + ps;
    }

    // C-layout -> A-fragment layout via wave-private LDS (intra-wave: in order)
#pragma unroll
    for (int r = 0; r < 8; ++r) {
      const int row = r + 8 * lh;
      Pl[wave][row][l16] = (bf16_t)sc[0][r];
      Pl[wave][row][16 + l16] = (bf16_t)sc[1][r];
    }
    V16 pf;
    pf.h[0] = *reinterpret_cast<const v8bf*>(&Pl[wave][l16][8 * lh]);
    pf.h[1] = *reinterpret_cast<const v8bf*>(&Pl[wave][l16][16 + 8 * lh]);

    // acc = acc*alpha + P @ V   (4 depth-groups of 16)
#pragma unroll
    for (int g = 0; g < 4; ++g) {
#pragma unroll
      for (int r = 0; r < 8; ++r) acc[g][r] *= alpha[r];
      V16 bv;  // B fragment: lane = depth col, K = key in table order (8*lh, 16+8*lh)
      bv.h[0] = *reinterpret_cast<const v8bf*>(&Vt[g * 16 + l16][8 * lh]);
      bv.h[1] = *reinterpret_cast<const v8bf*>(&Vt[g * 16 + l16][16 + 8 * lh]);
      acc[g] = wmma_bf16(pf.v, bv.v, acc[g]);
    }
  }

  // normalize and write concat-heads bf16 [B,S,H*64]
#pragma unroll
  for (int g = 0; g < 4; ++g) {
#pragma unroll
    for (int r = 0; r < 8; ++r) {
      const int srow = q0 + r + 8 * lh;
      const int feat = h * DEP + g * 16 + l16;
      O[((size_t)b * SEQ + srow) * DM + feat] = (bf16_t)(acc[g][r] / lrow[r]);
    }
  }
}

// ---------------------------------------------------------------------------
extern "C" void kernel_launch(void* const* d_in, const int* in_sizes, int n_in,
                              void* d_out, int out_size, void* d_ws, size_t ws_size,
                              hipStream_t stream) {
  const float* q    = (const float*)d_in[0];
  const float* k    = (const float*)d_in[1];
  const float* v    = (const float*)d_in[2];
  const float* mask = (const float*)d_in[3];
  const float* wq   = (const float*)d_in[4];
  const float* wk   = (const float*)d_in[5];
  const float* wv   = (const float*)d_in[6];
  const float* wo   = (const float*)d_in[7];
  float* out = (float*)d_out;

  const size_t elems = (size_t)MTOT * DM;  // 4M bf16 per tensor
  bf16_t* Qh = (bf16_t*)d_ws;
  bf16_t* Kh = Qh + elems;
  bf16_t* Vh = Kh + elems;
  bf16_t* Ao = Vh + elems;

  dim3 gg(MTOT / 128, DM / 128);
  gemm_qkv<<<gg, 256, 0, stream>>>(q, wq, Qh);
  gemm_qkv<<<gg, 256, 0, stream>>>(k, wk, Kh);
  gemm_qkv<<<gg, 256, 0, stream>>>(v, wv, Vh);
  attn<<<dim3(SEQ / 128, NH, NB), 256, 0, stream>>>(Qh, Kh, Vh, mask, Ao);
  gemm_out<<<gg, 256, 0, stream>>>(Ao, wo, out);
}
